// SkeletonGCN_45732811767949
// MI455X (gfx1250) — matrix operand order repaired and checked by reading, not compiled, not run on hardware
//
#include <hip/hip_runtime.h>
#include <hip/hip_bf16.h>
#include <hip/hip_fp16.h>

// ---------------- problem constants ----------------
#define J      25
#define BATCH  64
#define NF     300
#define HDIM   64
#define NCLS   60
#define NFRM   (BATCH * NF)     // 19200
#define NROWS  (NFRM * J)       // 480000 (divisible by 64)
#define C1IN   (J * HDIM)       // 1600
#define C1OUT  128
#define C2OUT  256

typedef _Float16 h16;
typedef __attribute__((ext_vector_type(8)))  _Float16 v8h;
typedef __attribute__((ext_vector_type(16))) _Float16 v16h;
typedef __attribute__((ext_vector_type(8)))  float    v8f;

#define WMMA_F16(a, b, c) \
  __builtin_amdgcn_wmma_f32_16x16x32_f16(false, (a), false, (b), (short)0, (c), false, false)

// ---- fragment gathers per CDNA5 ISA 7.12.2 VGPR layouts (wave32) ----
// A: 16(M) x 32(K) f16, row-major LDS tile, leading dim `ld` (halfs; 2*ld mult of 16B)
__device__ __forceinline__ v16h frag_a(const h16* p, int ld, int lane) {
  const int g = lane >> 4, m = lane & 15;
  const h16* base = p + m * ld + (g << 3);
  v8h lo = *(const v8h*)(base);         // K 8g..8g+7
  v8h hi = *(const v8h*)(base + 16);    // K 8g+16..8g+23
  return __builtin_shufflevector(lo, hi, 0,1,2,3,4,5,6,7,8,9,10,11,12,13,14,15);
}
// B: 32(K) x 16(N) f16 gathered from TRANSPOSED LDS tile pT[n][k], leading dim `ld`.
__device__ __forceinline__ v16h frag_bT(const h16* pT, int ld, int lane) {
  const int g = lane >> 4, n = lane & 15;
  const h16* base = pT + n * ld + (g << 3);
  v8h lo = *(const v8h*)(base);
  v8h hi = *(const v8h*)(base + 16);
  return __builtin_shufflevector(lo, hi, 0,1,2,3,4,5,6,7,8,9,10,11,12,13,14,15);
}
// D (f32): element r -> M = r + 8*(lane>>4), N = lane&15

// ---------------- K0a: pack conv weight (O, C, 3) f32 -> (O, 3, C) half ----------------
__global__ void k_pack_conv(const float* __restrict__ w, h16* __restrict__ out, int O, int Cin) {
  const int total = O * Cin * 3;
  for (int i = blockIdx.x * blockDim.x + threadIdx.x; i < total; i += gridDim.x * blockDim.x) {
    int o = i / (Cin * 3);
    int r = i % (Cin * 3);
    int k = r / Cin, c = r % Cin;
    out[i] = (h16)w[(unsigned)o * Cin * 3 + (unsigned)c * 3 + k];
  }
}

// ---------------- K0b: pack w2 (64x64) f32 -> transposed half w2t[n*64+k] ----------------
__global__ void k_pack_w2(const float* __restrict__ w2, h16* __restrict__ w2t) {
  const int i = blockIdx.x * 64 + threadIdx.x;   // grid 64, block 64
  const int k = i >> 6, n = i & 63;
  w2t[n * 64 + k] = (h16)w2[i];
}

// ---------------- K1: A_hat = D^-1/2 (A + I) D^-1/2 ----------------
__global__ void k_ahat(const int* __restrict__ ei, float* __restrict__ ahat) {
  __shared__ float A[J * J];
  __shared__ float dinv[J];
  const int tid = threadIdx.x;                      // 32 threads
  for (int i = tid; i < J * J; i += 32) A[i] = 0.f;
  __syncthreads();
  if (tid == 0) {
    for (int e = 0; e < 48; ++e) {                  // edge_index row0=src, row1=dst
      int s = ei[e], d = ei[48 + e];
      if (s >= 0 && s < J && d >= 0 && d < J) A[d * J + s] += 1.f;
    }
    for (int i = 0; i < J; ++i) A[i * J + i] += 1.f;
  }
  __syncthreads();
  if (tid < J) {
    float deg = 0.f;
    for (int c = 0; c < J; ++c) deg += A[tid * J + c];
    dinv[tid] = (deg > 0.f) ? rsqrtf(deg) : 0.f;
  }
  __syncthreads();
  for (int i = tid; i < J * J; i += 32) {
    int r = i / J, c = i % J;
    ahat[i] = dinv[r] * A[i] * dinv[c];
  }
}

// ---------------- K2: h1 = relu((A_hat @ x) @ w1 + b1), per frame ----------------
__global__ void k_gcn1(const float* __restrict__ x, const float* __restrict__ ahat,
                       const float* __restrict__ w1, const float* __restrict__ b1,
                       h16* __restrict__ h1) {
  __shared__ float As[J * J];
  __shared__ float xs[J * 3];
  __shared__ float ax[J * 3];
  const unsigned n = blockIdx.x;
  const int tid = threadIdx.x;                      // 128 threads
  for (int i = tid; i < J * J; i += 128) As[i] = ahat[i];
  for (int i = tid; i < J * 3; i += 128) xs[i] = x[n * (J * 3u) + i];
  __syncthreads();
  if (tid < J * 3) {
    int j = tid / 3, d = tid % 3;
    float s = 0.f;
    for (int i = 0; i < J; ++i) s += As[j * J + i] * xs[i * 3 + d];
    ax[tid] = s;
  }
  __syncthreads();
  for (int idx = tid; idx < J * HDIM; idx += 128) {
    int j = idx >> 6, h = idx & 63;
    float v = b1[h] + ax[j * 3 + 0] * w1[h] + ax[j * 3 + 1] * w1[64 + h]
                    + ax[j * 3 + 2] * w1[128 + h];
    h1[(n * J + j) * (unsigned)HDIM + h] = (h16)fmaxf(v, 0.f);
  }
}

// ---------------- K3: P = A_hat @ h1, per frame ----------------
__global__ void k_gcn2a(const h16* __restrict__ h1, const float* __restrict__ ahat,
                        h16* __restrict__ p) {
  __shared__ float As[J * J];
  __shared__ float hs[J * HDIM];
  const unsigned n = blockIdx.x;
  const int tid = threadIdx.x;                      // 128 threads
  for (int i = tid; i < J * J; i += 128) As[i] = ahat[i];
  for (int i = tid; i < J * HDIM; i += 128) hs[i] = (float)h1[n * (J * HDIM) + i];
  __syncthreads();
  for (int idx = tid; idx < J * HDIM; idx += 128) {
    int j = idx >> 6, h = idx & 63;
    float s = 0.f;
    for (int i = 0; i < J; ++i) s += As[j * J + i] * hs[i * HDIM + h];
    p[n * (J * HDIM) + idx] = (h16)s;
  }
}

// ---------------- K4: WMMA GEMM h2 = relu(P @ w2 + b2), scatter to (B, J*H, F) ----------------
// 480000 x 64 x 64; block = 4 waves, 64 rows; LDS strides padded to 72 halfs.
__global__ void k_gemm_w2(const h16* __restrict__ p, const h16* __restrict__ w2t,
                          const float* __restrict__ b2, h16* __restrict__ xc) {
  __shared__ __align__(16) h16 w2s[HDIM * 72];   // transposed [n][k]
  __shared__ __align__(16) h16 as[64 * 72];      // [row][k]
  const int tid = threadIdx.x;       // 128 threads = 4 waves
  const int lane = tid & 31, wv = tid >> 5;
  const unsigned rowBase = blockIdx.x * 64u;
  for (int i = tid; i < HDIM * HDIM; i += 128) {
    int n = i >> 6, k = i & 63;
    w2s[n * 72 + k] = w2t[i];
  }
  for (int i = tid; i < 64 * HDIM; i += 128) {
    int r = i >> 6, k = i & 63;
    as[r * 72 + k] = p[(rowBase + r) * (unsigned)HDIM + k];
  }
  __syncthreads();

  v8f acc[4] = {};
#pragma unroll
  for (int kk = 0; kk < HDIM; kk += 32) {
    v16h a = frag_a(&as[(wv * 16) * 72 + kk], 72, lane);
#pragma unroll
    for (int nt = 0; nt < 4; ++nt) {
      v16h b = frag_bT(&w2s[(nt * 16) * 72 + kk], 72, lane);
      acc[nt] = WMMA_F16(a, b, acc[nt]);
    }
  }

  const int g = lane >> 4, nn = lane & 15;
#pragma unroll
  for (int nt = 0; nt < 4; ++nt) {
    int h = nt * 16 + nn;
    float bias = b2[h];
#pragma unroll
    for (int r = 0; r < 8; ++r) {
      unsigned M  = rowBase + (unsigned)(wv * 16 + r + 8 * g);   // < 480000, 32-bit math
      float v = fmaxf(acc[nt][r] + bias, 0.f);
      unsigned nf = M / J,  jj = M - nf * J;
      unsigned bb = nf / NF, f = nf - bb * NF;
      xc[(bb * C1IN + jj * HDIM + h) * (unsigned)NF + f] = (h16)v;
    }
  }
}

// ---------------- K5: conv1 as WMMA im2col GEMM; block: M=128 x N=64, K=4800 ----------------
// A tile caches all 3 taps per K-chunk; one raw 66x32 B tile serves all taps (shifted reads).
__global__ void k_conv1(const h16* __restrict__ xc, const h16* __restrict__ wp,
                        const float* __restrict__ bias, h16* __restrict__ y1) {
  __shared__ __align__(16) h16 as[C1OUT * 120];   // [o][tap*40 + kk]   30720 B
  __shared__ __align__(16) h16 bt[66 * 40];       // [t'][kk], t' = st-(tbase-1): 0..65
  const int tid = threadIdx.x;       // 256 threads = 8 waves
  const int lane = tid & 31, wv = tid >> 5;
  const int b = blockIdx.y, tbase = blockIdx.x * 64;
  v8f acc[4] = {};

  for (int c0 = 0; c0 < C1IN; c0 += 32) {
    // A: 128 rows x 3 taps x 32 halfs as aligned v8h chunks (1536 chunks, 6/thread)
    for (int ch = tid; ch < 1536; ch += 256) {
      int o = ch / 12, rem = ch % 12;
      int k = rem >> 2, c8 = (rem & 3) << 3;
      const h16* gp = &wp[(unsigned)o * (C1IN * 3) + (unsigned)k * C1IN + c0 + c8];
      *(v8h*)&as[o * 120 + k * 40 + c8] = *(const v8h*)gp;
      __builtin_prefetch(gp + 32, 0, 0);          // next K-chunk (global_prefetch_b8)
    }
    // B raw tile: st = tbase-1 .. tbase+64, zero-padded
    for (int i = tid; i < 66 * 32; i += 256) {
      int tt = i % 66, kk = i / 66;               // tt fast -> coalesced
      int st = tbase - 1 + tt;
      h16 v = (h16)0.f;
      if (st >= 0 && st < NF)
        v = xc[((unsigned)b * C1IN + c0 + kk) * NF + st];
      bt[tt * 40 + kk] = v;
    }
    __syncthreads();
#pragma unroll
    for (int k = 0; k < 3; ++k) {
      v16h a = frag_a(&as[(wv * 16) * 120 + k * 40], 120, lane);
#pragma unroll
      for (int sub = 0; sub < 4; ++sub)
        acc[sub] = WMMA_F16(a, frag_bT(&bt[(sub * 16 + k) * 40], 40, lane), acc[sub]);
    }
    __syncthreads();
  }

  const int g = lane >> 4, nn = lane & 15;
#pragma unroll
  for (int sub = 0; sub < 4; ++sub) {
#pragma unroll
    for (int r = 0; r < 8; ++r) {
      int o = wv * 16 + r + 8 * g;
      int t = tbase + sub * 16 + nn;
      if (t < NF)
        y1[((unsigned)b * C1OUT + o) * NF + t] = (h16)fmaxf(acc[sub][r] + bias[o], 0.f);
    }
  }
}

// ---------------- K6: conv2 (WMMA) + ReLU + mean-pool (deterministic LDS reduce) ----------------
__global__ void k_conv2_pool(const h16* __restrict__ y1, const h16* __restrict__ wp,
                             const float* __restrict__ bias, float* __restrict__ pool) {
  __shared__ __align__(16) h16  as[128 * 120];    // [o][tap*40 + kk]
  __shared__ __align__(16) h16  bt[66 * 40];
  __shared__ float red[128 * 16];
  const int tid = threadIdx.x;       // 256 threads = 8 waves
  const int lane = tid & 31, wv = tid >> 5;
  const int b = blockIdx.y, mh = blockIdx.x;      // mh in {0,1}: which 128 of 256 outputs
  const int g = lane >> 4, nn = lane & 15;
  float psum[8];
#pragma unroll
  for (int r = 0; r < 8; ++r) psum[r] = 0.f;

  for (int tile = 0; tile < 5; ++tile) {          // 5 * 64 = 320 >= 300
    const int tbase = tile * 64;
    v8f acc[4] = {};
    for (int c0 = 0; c0 < C1OUT; c0 += 32) {
      for (int ch = tid; ch < 1536; ch += 256) {
        int o = ch / 12, rem = ch % 12;
        int k = rem >> 2, c8 = (rem & 3) << 3;
        const h16* gp = &wp[(unsigned)(o + mh * 128) * (C1OUT * 3) + (unsigned)k * C1OUT + c0 + c8];
        *(v8h*)&as[o * 120 + k * 40 + c8] = *(const v8h*)gp;
      }
      for (int i = tid; i < 66 * 32; i += 256) {
        int tt = i % 66, kk = i / 66;
        int st = tbase - 1 + tt;
        h16 v = (h16)0.f;
        if (st >= 0 && st < NF)
          v = y1[((unsigned)b * C1OUT + c0 + kk) * NF + st];
        bt[tt * 40 + kk] = v;
      }
      __syncthreads();
#pragma unroll
      for (int k = 0; k < 3; ++k) {
        v16h a = frag_a(&as[(wv * 16) * 120 + k * 40], 120, lane);
#pragma unroll
        for (int sub = 0; sub < 4; ++sub)
          acc[sub] = WMMA_F16(a, frag_bT(&bt[(sub * 16 + k) * 40], 40, lane), acc[sub]);
      }
      __syncthreads();
    }
#pragma unroll
    for (int sub = 0; sub < 4; ++sub) {
#pragma unroll
      for (int r = 0; r < 8; ++r) {
        int o = mh * 128 + wv * 16 + r + 8 * g;
        int t = tbase + sub * 16 + nn;
        if (t < NF) psum[r] += fmaxf(acc[sub][r] + bias[o], 0.f);
      }
    }
  }
#pragma unroll
  for (int r = 0; r < 8; ++r)
    red[(wv * 16 + r + 8 * g) * 16 + nn] = psum[r];
  __syncthreads();
  if (tid < 128) {
    float s = 0.f;
    for (int c = 0; c < 16; ++c) s += red[tid * 16 + c];
    pool[(unsigned)b * C2OUT + mh * 128 + tid] = s * (1.0f / NF);
  }
}

// ---------------- K7: out = sigmoid(pool @ fc_w.T + fc_b) ----------------
__global__ void k_fc(const float* __restrict__ pool, const float* __restrict__ fw,
                     const float* __restrict__ fb, float* __restrict__ out) {
  const int b = blockIdx.x, t = threadIdx.x;      // 64 threads
  if (t >= NCLS) return;
  float s = fb[t];
  for (int o = 0; o < C2OUT; ++o) s += pool[(unsigned)b * C2OUT + o] * fw[(unsigned)t * C2OUT + o];
  out[(unsigned)b * NCLS + t] = 1.f / (1.f + __expf(-s));
}

// ---------------- launcher ----------------
extern "C" void kernel_launch(void* const* d_in, const int* in_sizes, int n_in,
                              void* d_out, int out_size, void* d_ws, size_t ws_size,
                              hipStream_t stream) {
  const float* x   = (const float*)d_in[0];
  const int*   ei  = (const int*)d_in[1];   // edge_index, 2x48 (harness int convention)
  const float* w1  = (const float*)d_in[4];
  const float* b1  = (const float*)d_in[5];
  const float* w2  = (const float*)d_in[6];
  const float* b2  = (const float*)d_in[7];
  const float* c1w = (const float*)d_in[8];
  const float* c1b = (const float*)d_in[9];
  const float* c2w = (const float*)d_in[10];
  const float* c2b = (const float*)d_in[11];
  const float* fw  = (const float*)d_in[12];
  const float* fb  = (const float*)d_in[13];

  // workspace carve-out (~130 MB total; all intermediates half precision)
  char* ws = (char*)d_ws;
  size_t off = 0;
  auto take = [&](size_t bytes) -> void* {
    void* p = (void*)(ws + off);
    off += (bytes + 255) & ~(size_t)255;
    return p;
  };
  float* ahat = (float*)take(J * J * sizeof(float));
  float* pool = (float*)take((size_t)BATCH * C2OUT * sizeof(float));
  h16*   w2t  = (h16*)take((size_t)HDIM * HDIM * sizeof(h16));             // 8 KB
  h16*   wp1  = (h16*)take((size_t)C1OUT * C1IN * 3 * sizeof(h16));        // 1.2 MB
  h16*   wp2  = (h16*)take((size_t)C2OUT * C1OUT * 3 * sizeof(h16));       // 196 KB
  h16*   y1   = (h16*)take((size_t)BATCH * C1OUT * NF * sizeof(h16));      // 4.9 MB
  h16*   bufA = (h16*)take((size_t)NROWS * HDIM * sizeof(h16));            // 61.4 MB (h1 -> Xc)
  h16*   bufB = (h16*)take((size_t)NROWS * HDIM * sizeof(h16));            // 61.4 MB (P)

  k_ahat<<<1, 32, 0, stream>>>(ei, ahat);
  k_pack_w2<<<64, 64, 0, stream>>>(w2, w2t);
  k_pack_conv<<<512, 256, 0, stream>>>(c1w, wp1, C1OUT, C1IN);
  k_pack_conv<<<128, 256, 0, stream>>>(c2w, wp2, C2OUT, C1OUT);
  k_gcn1<<<NFRM, 128, 0, stream>>>(x, ahat, w1, b1, bufA);
  k_gcn2a<<<NFRM, 128, 0, stream>>>(bufA, ahat, bufB);
  k_gemm_w2<<<NROWS / 64, 128, 0, stream>>>(bufB, w2t, b2, bufA);          // bufA reused as Xc
  k_conv1<<<dim3(5, BATCH), 256, 0, stream>>>(bufA, wp1, c1b, y1);
  k_conv2_pool<<<dim3(2, BATCH), 256, 0, stream>>>(y1, wp2, c2b, pool);
  k_fc<<<BATCH, 64, 0, stream>>>(pool, fw, fb, (float*)d_out);
}
